// ExecPolicyNetwork_12979391169443
// MI455X (gfx1250) — matrix operand order repaired and checked by reading, not compiled, not run on hardware
//
#include <hip/hip_runtime.h>

typedef _Float16 v16h __attribute__((ext_vector_type(16)));
typedef _Float16 v8h  __attribute__((ext_vector_type(8)));
typedef _Float16 v4h  __attribute__((ext_vector_type(4)));
typedef float    v8f  __attribute__((ext_vector_type(8)));

#define NFRAG 21
#define FEAT_STRIDE 40   // halfs per staged feature row (36 real + 4 pad), 80B = 16B aligned

__device__ __forceinline__ v8f wmma16(v16h a, v16h b, v8f c) {
  // (neg_a, A, neg_b, B, c_mod, C, reuse_a, reuse_b)
  return __builtin_amdgcn_wmma_f32_16x16x32_f16(false, a, false, b, (short)0, c, false, false);
}
__device__ __forceinline__ v16h cat8(v8h a, v8h b) {
  v16h r;
#pragma unroll
  for (int i = 0; i < 8; ++i) { r[i] = a[i]; r[i + 8] = b[i]; }
  return r;
}
__device__ __forceinline__ v8f zero8f() {
  v8f r;
#pragma unroll
  for (int i = 0; i < 8; ++i) r[i] = 0.0f;
  return r;
}
__device__ __forceinline__ v16h zero16h() {
  v16h r;
#pragma unroll
  for (int i = 0; i < 16; ++i) r[i] = (_Float16)0;
  return r;
}

// ---------------- Kernel 1: exclusive prefix sum of n_sel over J jobs -------
__global__ void scan_kernel(const int* __restrict__ jobidx,
                            const int* __restrict__ nacts,
                            int* __restrict__ starts, int J) {
  __shared__ int part[1024];
  const int tid = threadIdx.x;
  const int chunk = (J + 1023) >> 10;
  const int beg = tid * chunk;
  const int end = (beg + chunk < J) ? beg + chunk : J;
  int s = 0;
  for (int i = beg; i < end; ++i) s += nacts[jobidx[i]];
  part[tid] = s;
  __syncthreads();
  for (int off = 1; off < 1024; off <<= 1) {
    int v = part[tid];
    int add = (tid >= off) ? part[tid - off] : 0;
    __syncthreads();
    part[tid] = v + add;
    __syncthreads();
  }
  int run = part[tid] - s;  // exclusive
  for (int i = beg; i < end; ++i) { starts[i] = run; run += nacts[jobidx[i]]; }
}

// ---------------- Kernel 2: rowq[t] = q (the reference's rpt) ---------------
__global__ void fill_rowq_kernel(const int* __restrict__ starts,
                                 const int* __restrict__ jobidx,
                                 const int* __restrict__ nacts,
                                 int* __restrict__ rowq, int J) {
  int q = blockIdx.x * blockDim.x + threadIdx.x;
  if (q >= J) return;
  int s = starts[q];
  int n = nacts[jobidx[q]];
  for (int k = 0; k < n; ++k) rowq[s + k] = q;
}

// ---------------- Kernel 3: stage 36-d features as f16 rows -----------------
__global__ void prep_feat_kernel(const float* __restrict__ x,
                                 const float* __restrict__ h_dag,
                                 const float* __restrict__ h_glob,
                                 const int* __restrict__ ptr,
                                 const int* __restrict__ jobidx,
                                 _Float16* __restrict__ featH, int J) {
  int q = blockIdx.x * blockDim.x + threadIdx.x;
  if (q >= J) return;
  int jid = jobidx[q];
  int node = ptr[jid];
  const float* xr = x + (size_t)node * 5;
  _Float16* o = featH + (size_t)q * FEAT_STRIDE;
#pragma unroll
  for (int e = 0; e < 3; ++e) o[e] = (_Float16)xr[e];
#pragma unroll
  for (int e = 0; e < 16; ++e) o[3 + e] = (_Float16)h_dag[(size_t)jid * 16 + e];
#pragma unroll
  for (int e = 0; e < 16; ++e) o[19 + e] = (_Float16)h_glob[(size_t)q * 16 + e];
#pragma unroll
  for (int e = 35; e < FEAT_STRIDE; ++e) o[e] = (_Float16)0;  // exec slot filled per row
}

// ---------------- Kernel 4: pre-swizzle weights into B-fragment layout ------
// frag f, lane, element i:  b[i] = W[c*32 + (i<8 ? hi*8+i : 16+hi*8+i-8)][nt*16 + l]  (0-padded)
__global__ void prep_weights_kernel(const float* __restrict__ W1,
                                    const float* __restrict__ W2,
                                    const float* __restrict__ W3,
                                    const float* __restrict__ W4,
                                    _Float16* __restrict__ wH) {
  int w = blockIdx.x * blockDim.x + threadIdx.x;
  if (w >= NFRAG * 32) return;
  int f = w >> 5, lane = w & 31, hi = lane >> 4, l = lane & 15;
  const float* src; int K, N, c, nt;
  if (f < 8)       { src = W1; K = 36; N = 64; c = f >> 2;        nt = f & 3; }
  else if (f < 16) { src = W2; K = 64; N = 64; c = (f - 8) >> 2;  nt = (f - 8) & 3; }
  else if (f < 20) { src = W3; K = 64; N = 32; c = (f - 16) >> 1; nt = (f - 16) & 1; }
  else             { src = W4; K = 32; N = 1;  c = 0;             nt = 0; }
  _Float16* dst = wH + (size_t)(f * 32 + lane) * 16;
  for (int i = 0; i < 16; ++i) {
    int k = c * 32 + (i < 8 ? hi * 8 + i : 16 + hi * 8 + (i - 8));
    int n = nt * 16 + l;
    float v = (k < K && n < N) ? src[(size_t)k * N + n] : 0.0f;
    dst[i] = (_Float16)v;
  }
}

// ---------------- Kernel 5: the WMMA MLP, one wave per 16-row tile ----------
__global__ void mlp_kernel(const _Float16* __restrict__ featH,
                           const int* __restrict__ rowq,
                           const int* __restrict__ execIdx,
                           const _Float16* __restrict__ wH,
                           const float* __restrict__ b1,
                           const float* __restrict__ b2,
                           const float* __restrict__ b3,
                           const float* __restrict__ b4,
                           float* __restrict__ out,
                           int T, int NT) {
  // Weights live in LDS (shared by all 8 waves); activations get a 2KB slice per wave.
  __shared__ __align__(16) _Float16 wsh[NFRAG * 32 * 16];  // 21504 B
  __shared__ __align__(16) _Float16 sm[8][16 * 64];        // 16384 B
  const int lane = threadIdx.x & 31;
  const int wid = threadIdx.x >> 5;
  const int l = lane & 15, hi = lane >> 4;
  _Float16* smw = &sm[wid][0];

  // Cooperative copy of the pre-swizzled weight fragments into LDS.
  {
    const uint4* src = (const uint4*)wH;
    uint4* dst = (uint4*)wsh;
    for (int i = threadIdx.x; i < (NFRAG * 32 * 16 * 2) / 16; i += blockDim.x)
      dst[i] = src[i];
  }
  __syncthreads();

  // Per-lane LDS base for fragment loads; each frag is base + f*1024B (folds
  // into the ds_load immediate offset). Loads are plain ds_load_b128; the
  // asm memory fences in the loop keep them from being hoisted out (which
  // would blow past 256 VGPRs and spill to scratch, as seen in round 1).
  const _Float16* wbase = wsh + (size_t)lane * 16;
  auto ldw = [&](int f) -> v16h {
    const v8h* p = (const v8h*)(wbase + (size_t)f * 32 * 16);
    return cat8(p[0], p[1]);
  };

  float b1v[4], b2v[4], b3v[2];
#pragma unroll
  for (int nt = 0; nt < 4; ++nt) { b1v[nt] = b1[nt * 16 + l]; b2v[nt] = b2[nt * 16 + l]; }
#pragma unroll
  for (int nt = 0; nt < 2; ++nt) { b3v[nt] = b3[nt * 16 + l]; }
  const float b4v = b4[0];

  const int gw = blockIdx.x * 8 + wid;
  const int gs = gridDim.x * 8;
  for (int tile = gw; tile < NT; tile += gs) {
    const int trow = tile * 16 + l;
    const int tcl = trow < T ? trow : T - 1;  // clamp tail; stores are guarded
    const int q = rowq[tcl];
    const float ev = (float)execIdx[tcl] * (1.0f / 50.0f);
    const _Float16* fr = featH + (size_t)q * FEAT_STRIDE;

    // Layer-1 A fragments per ISA 16-bit A layout (lanes 0-15: K 0-7 & 16-23; 16-31: K 8-15 & 24-31)
    v16h a0 = cat8(*(const v8h*)(fr + hi * 8), *(const v8h*)(fr + 16 + hi * 8));
    v16h a1 = zero16h();  // K=32..63 chunk: only K=32..35 real (lanes 0-15)
    if (hi == 0) {
      v4h t4 = *(const v4h*)(fr + 32);
      a1[0] = t4[0]; a1[1] = t4[1]; a1[2] = t4[2]; a1[3] = (_Float16)ev;
    }
    // ---- Layer 1: 36(->64) x 64 ----
#pragma unroll
    for (int nt = 0; nt < 4; ++nt) {
      v8f c = zero8f();
      c = wmma16(a0, ldw(nt), c);
      c = wmma16(a1, ldw(4 + nt), c);
#pragma unroll
      for (int r = 0; r < 8; ++r) {
        float v = c[r] + b1v[nt];
        v = v > 0.0f ? v : 0.0f;
        smw[(r + hi * 8) * 64 + nt * 16 + l] = (_Float16)v;  // row-major [16][64]
      }
    }
    asm volatile("s_wait_dscnt 0" ::: "memory");
    // ---- Layer 2: 64 x 64 ----
    v16h a2c0 = cat8(*(const v8h*)(smw + l * 64 + hi * 8),
                     *(const v8h*)(smw + l * 64 + 16 + hi * 8));
    v16h a2c1 = cat8(*(const v8h*)(smw + l * 64 + 32 + hi * 8),
                     *(const v8h*)(smw + l * 64 + 48 + hi * 8));
#pragma unroll
    for (int nt = 0; nt < 4; ++nt) {
      v8f c = zero8f();
      c = wmma16(a2c0, ldw(8 + nt), c);
      c = wmma16(a2c1, ldw(12 + nt), c);
#pragma unroll
      for (int r = 0; r < 8; ++r) {
        float v = c[r] + b2v[nt];
        v = v > 0.0f ? v : 0.0f;
        smw[(r + hi * 8) * 64 + nt * 16 + l] = (_Float16)v;
      }
    }
    asm volatile("s_wait_dscnt 0" ::: "memory");
    // ---- Layer 3: 64 x 32 ----
    v16h a3c0 = cat8(*(const v8h*)(smw + l * 64 + hi * 8),
                     *(const v8h*)(smw + l * 64 + 16 + hi * 8));
    v16h a3c1 = cat8(*(const v8h*)(smw + l * 64 + 32 + hi * 8),
                     *(const v8h*)(smw + l * 64 + 48 + hi * 8));
#pragma unroll
    for (int nt = 0; nt < 2; ++nt) {
      v8f c = zero8f();
      c = wmma16(a3c0, ldw(16 + nt), c);
      c = wmma16(a3c1, ldw(18 + nt), c);
#pragma unroll
      for (int r = 0; r < 8; ++r) {
        float v = c[r] + b3v[nt];
        v = v > 0.0f ? v : 0.0f;
        smw[(r + hi * 8) * 32 + nt * 16 + l] = (_Float16)v;  // row-major [16][32]
      }
    }
    asm volatile("s_wait_dscnt 0" ::: "memory");
    // ---- Layer 4: 32 x 1 (W4 in column 0 of a 32x16 B fragment) ----
    v16h a4 = cat8(*(const v8h*)(smw + l * 32 + hi * 8),
                   *(const v8h*)(smw + l * 32 + 16 + hi * 8));
    v8f d = wmma16(a4, ldw(20), zero8f());
    if (l == 0) {  // lanes holding N==0: lane0 -> M 0..7, lane16 -> M 8..15
#pragma unroll
      for (int r = 0; r < 8; ++r) {
        int t = tile * 16 + hi * 8 + r;
        if (t < T) out[t] = d[r] + b4v;
      }
    }
    asm volatile("s_wait_dscnt 0" ::: "memory");  // drain before next iter reuses sm
  }
}

// ---------------------------------------------------------------------------
extern "C" void kernel_launch(void* const* d_in, const int* in_sizes, int n_in,
                              void* d_out, int out_size, void* d_ws, size_t ws_size,
                              hipStream_t stream) {
  const float* x       = (const float*)d_in[0];
  const float* h_dag   = (const float*)d_in[1];
  const float* h_glob  = (const float*)d_in[2];
  const int*   ptr     = (const int*)d_in[3];
  const int*   jobidx  = (const int*)d_in[4];
  const int*   nacts   = (const int*)d_in[5];
  const int*   execIdx = (const int*)d_in[6];
  const float* W1 = (const float*)d_in[7];
  const float* b1 = (const float*)d_in[8];
  const float* W2 = (const float*)d_in[9];
  const float* b2 = (const float*)d_in[10];
  const float* W3 = (const float*)d_in[11];
  const float* b3 = (const float*)d_in[12];
  const float* W4 = (const float*)d_in[13];
  const float* b4 = (const float*)d_in[14];
  float* out = (float*)d_out;

  const int J = in_sizes[4];   // number of jobs (== B)
  const int T = in_sizes[6];   // total ragged rows (== out_size)

  auto align16 = [](size_t v) { return (v + 15) & ~(size_t)15; };
  char* w = (char*)d_ws;
  int* starts = (int*)w;                 w += align16((size_t)J * 4);
  int* rowq = (int*)w;                   w += align16((size_t)T * 4);
  _Float16* featH = (_Float16*)w;        w += align16((size_t)J * FEAT_STRIDE * 2);
  _Float16* wH = (_Float16*)w;           // NFRAG*32*16 halfs = 21504 B
  (void)ws_size; (void)n_in; (void)out_size;

  scan_kernel<<<1, 1024, 0, stream>>>(jobidx, nacts, starts, J);
  fill_rowq_kernel<<<(J + 255) / 256, 256, 0, stream>>>(starts, jobidx, nacts, rowq, J);
  prep_feat_kernel<<<(J + 255) / 256, 256, 0, stream>>>(x, h_dag, h_glob, ptr, jobidx, featH, J);
  prep_weights_kernel<<<3, 256, 0, stream>>>(W1, W2, W3, W4, wH);

  const int NT = (T + 15) / 16;
  int blocks = (NT + 7) / 8;
  if (blocks > 3072) blocks = 3072;
  if (blocks < 1) blocks = 1;
  mlp_kernel<<<blocks, 256, 0, stream>>>(featH, rowq, execIdx, wH, b1, b2, b3, b4, out, T, NT);
}
